// SelfMultiheadAttn_39702677684466
// MI455X (gfx1250) — compile-verified, matrix operand order checked by reading, not run on hardware
//
#include <hip/hip_runtime.h>
#include <hip/hip_bf16.h>

#define S_LEN 2048
#define BATCH 4
#define EMB   1024
#define HEADS 16
#define HD    64
#define F3    3072
#define SBE   (S_LEN * BATCH * EMB)   // 8,388,608 elements

typedef __attribute__((ext_vector_type(16))) __bf16 v16bf;
typedef __attribute__((ext_vector_type(8)))  __bf16 v8bf;
typedef __attribute__((ext_vector_type(8)))  float  v8f;

// ---- fragment helpers -------------------------------------------------------

__device__ __forceinline__ v8bf load8bf(const __bf16* p) {
    union { uint4 u; v8bf v; } c;
    c.u = *reinterpret_cast<const uint4*>(p);
    return c.v;
}

__device__ __forceinline__ v8bf cvt8(const float* p) {
    float4 a = reinterpret_cast<const float4*>(p)[0];
    float4 b = reinterpret_cast<const float4*>(p)[1];
    v8bf r;
    r[0] = (__bf16)a.x; r[1] = (__bf16)a.y; r[2] = (__bf16)a.z; r[3] = (__bf16)a.w;
    r[4] = (__bf16)b.x; r[5] = (__bf16)b.y; r[6] = (__bf16)b.z; r[7] = (__bf16)b.w;
    return r;
}

__device__ __forceinline__ v16bf cat16(v8bf lo, v8bf hi) {
    return __builtin_shufflevector(lo, hi, 0,1,2,3,4,5,6,7,8,9,10,11,12,13,14,15);
}

// A/B fragment from row-major fp32 (convert) or bf16 source; per-lane row fixed.
__device__ __forceinline__ v16bf frag_f32(const float* row, int k0, int half) {
    return cat16(cvt8(row + k0 + half * 8), cvt8(row + k0 + 16 + half * 8));
}
__device__ __forceinline__ v16bf frag_bf16(const __bf16* row, int k0, int half) {
    return cat16(load8bf(row + k0 + half * 8), load8bf(row + k0 + 16 + half * 8));
}

__device__ __forceinline__ v8f wmma_bf16(v16bf a, v16bf b, v8f c) {
    return __builtin_amdgcn_wmma_f32_16x16x32_bf16(
        /*neg_a=*/false, a, /*neg_b=*/false, b,
        /*c_mod=*/(short)0, c, /*reuse_a=*/false, /*reuse_b=*/false);
}

// lane-private LDS byte offset (low 32 bits of the flat LDS-aperture address)
__device__ __forceinline__ unsigned lds_off(const void* p) {
    return (unsigned)(uintptr_t)p;
}

// ---- kernel 1: fused QKV projection ----------------------------------------
// lin[sb, f] = sum_e X[sb,e] * W[f,e] + bias[f]
// f = h*192 + which*64 + d ; scatter into Q/K/V [b,h,s,d] bf16 (scale folded into Q).

__device__ __forceinline__ void qkv_store8(
    __bf16* Qw, __bf16* Kw, __bf16* Vw, const float* bias,
    int f, int mrow_base, int half, const v8f& acc)
{
    const int h     = f / 192;
    const int rem   = f - h * 192;
    const int which = rem >> 6;
    const int d     = rem & 63;
    const float bv  = bias[f];
    __bf16* dst = (which == 0) ? Qw : ((which == 1) ? Kw : Vw);
    const float mult = (which == 0) ? 0.125f : 1.0f;   // hd^-0.5 folded into Q
#pragma unroll
    for (int r = 0; r < 8; ++r) {
        const int row = mrow_base + r + 8 * half;      // row = s*B + b
        const int s = row >> 2, bb = row & 3;
        dst[((size_t)(bb * HEADS + h) * S_LEN + s) * HD + d] =
            (__bf16)((acc[r] + bv) * mult);
    }
}

__global__ __launch_bounds__(256) void qkv_proj_kernel(
    const float* __restrict__ Xq, const float* __restrict__ W,
    const float* __restrict__ bias,
    unsigned short* __restrict__ Qw_, unsigned short* __restrict__ Kw_,
    unsigned short* __restrict__ Vw_)
{
    __bf16* Qw = reinterpret_cast<__bf16*>(Qw_);
    __bf16* Kw = reinterpret_cast<__bf16*>(Kw_);
    __bf16* Vw = reinterpret_cast<__bf16*>(Vw_);

    const int lane = threadIdx.x & 31;
    const int wave = threadIdx.x >> 5;
    const int half = lane >> 4;
    const int mn   = lane & 15;
    const int m0 = blockIdx.x * 64  + (wave & 1) * 32;   // wave tile: 32M x 32N
    const int n0 = blockIdx.y * 128 + (wave >> 1) * 32;

    const float* arow0 = Xq + (size_t)(m0 + mn) * EMB;
    const float* arow1 = Xq + (size_t)(m0 + 16 + mn) * EMB;
    const float* brow0 = W  + (size_t)(n0 + mn) * EMB;
    const float* brow1 = W  + (size_t)(n0 + 16 + mn) * EMB;

    v8f acc00 = {}, acc01 = {}, acc10 = {}, acc11 = {};
    for (int k0 = 0; k0 < EMB; k0 += 32) {
        v16bf a0 = frag_f32(arow0, k0, half);
        v16bf a1 = frag_f32(arow1, k0, half);
        v16bf b0 = frag_f32(brow0, k0, half);
        v16bf b1 = frag_f32(brow1, k0, half);
        acc00 = wmma_bf16(a0, b0, acc00);
        acc01 = wmma_bf16(a0, b1, acc01);
        acc10 = wmma_bf16(a1, b0, acc10);
        acc11 = wmma_bf16(a1, b1, acc11);
    }

    qkv_store8(Qw, Kw, Vw, bias, n0 + mn,      m0,      half, acc00);
    qkv_store8(Qw, Kw, Vw, bias, n0 + 16 + mn, m0,      half, acc01);
    qkv_store8(Qw, Kw, Vw, bias, n0 + mn,      m0 + 16, half, acc10);
    qkv_store8(Qw, Kw, Vw, bias, n0 + 16 + mn, m0 + 16, half, acc11);
}

// ---- kernel 2: flash attention per (b,h) -----------------------------------
__global__ __launch_bounds__(128) void attn_kernel(
    const unsigned short* __restrict__ Qw_, const unsigned short* __restrict__ Kw_,
    const unsigned short* __restrict__ Vw_, unsigned short* __restrict__ CTX_)
{
    const __bf16* Qw = reinterpret_cast<const __bf16*>(Qw_);
    const __bf16* Kw = reinterpret_cast<const __bf16*>(Kw_);
    const __bf16* Vw = reinterpret_cast<const __bf16*>(Vw_);
    __bf16* CTX = reinterpret_cast<__bf16*>(CTX_);

    __shared__ __bf16 Klds[32 * 64];     // [key][d]
    __shared__ __bf16 VldsT[64 * 32];    // [d][key]
    __shared__ __bf16 Plds[4 * 16 * 32]; // per-wave P scratch

    const int tid  = threadIdx.x;
    const int lane = tid & 31;
    const int wave = tid >> 5;
    const int half = lane >> 4;
    const int mn   = lane & 15;

    const int bh = blockIdx.y;
    const int b  = bh >> 4;
    const int h  = bh & 15;
    const int qbase = blockIdx.x * 64;

    const __bf16* Qg = Qw + (size_t)bh * S_LEN * HD;
    const __bf16* Kg = Kw + (size_t)bh * S_LEN * HD;
    const __bf16* Vg = Vw + (size_t)bh * S_LEN * HD;

    // per-wave Q fragments (16 rows x 64), scale pre-applied
    const __bf16* qrow = Qg + (size_t)(qbase + wave * 16 + mn) * HD;
    const v16bf qa0 = frag_bf16(qrow, 0, half);
    const v16bf qa1 = frag_bf16(qrow, 32, half);

    v8f O[4] = {{}, {}, {}, {}};
    float m_r[8], l_r[8];
#pragma unroll
    for (int r = 0; r < 8; ++r) { m_r[r] = -3.0e38f; l_r[r] = 0.0f; }

    __bf16* P = Plds + wave * 16 * 32;

    for (int kb = 0; kb < S_LEN; kb += 32) {
        __syncthreads();
        // stage K tile (32x64 bf16 contiguous = 256 * 16B) via CDNA5 async copy:
        // global -> LDS direct, no VGPR round trip, tracked by ASYNCcnt.
        {
            const __bf16* src = Kg + (size_t)kb * HD;
            for (int i = tid; i < 256; i += 128) {
                unsigned dst = lds_off(Klds + i * 8);
                const void* g = (const void*)(src + i * 8);
                asm volatile("global_load_async_to_lds_b128 %0, %1, off"
                             :: "v"(dst), "v"(g) : "memory");
            }
        }
        // stage V tile transposed: VldsT[d][key] (transpose must pass through VGPRs)
        {
            const int kt = tid & 31;
            const int dg = tid >> 5;  // 0..3 -> 16 d each
            const __bf16* vp = Vg + (size_t)(kb + kt) * HD + dg * 16;
            v8bf va = load8bf(vp);
            v8bf vb = load8bf(vp + 8);
#pragma unroll
            for (int j = 0; j < 8; ++j) {
                VldsT[(dg * 16 + j) * 32 + kt]     = va[j];
                VldsT[(dg * 16 + 8 + j) * 32 + kt] = vb[j];
            }
        }
        if (kb + 32 < S_LEN)
            __builtin_prefetch(Kg + (size_t)(kb + 32) * HD + tid * 16, 0, 3);
        asm volatile("s_wait_asynccnt 0" ::: "memory");
        __syncthreads();

        // scores S = Q * K^T for 32 keys (two 16x16 tiles), K-dim = hd = 64
        v8f s0 = {}, s1 = {};
        {
            const __bf16* k0p = Klds + mn * HD;
            const __bf16* k1p = Klds + (16 + mn) * HD;
            s0 = wmma_bf16(qa0, frag_bf16(k0p, 0, half),  s0);
            s0 = wmma_bf16(qa1, frag_bf16(k0p, 32, half), s0);
            s1 = wmma_bf16(qa0, frag_bf16(k1p, 0, half),  s1);
            s1 = wmma_bf16(qa1, frag_bf16(k1p, 32, half), s1);
        }

        // online softmax (row stats replicated across the 16-lane half)
        float alpha[8];
#pragma unroll
        for (int r = 0; r < 8; ++r) {
            float mx = fmaxf(s0[r], s1[r]);
#pragma unroll
            for (int off = 1; off < 16; off <<= 1)
                mx = fmaxf(mx, __shfl_xor(mx, off, 32));
            const float mnew = fmaxf(m_r[r], mx);
            alpha[r] = exp2f((m_r[r] - mnew) * 1.44269504f);
            const float p0 = exp2f((s0[r] - mnew) * 1.44269504f);
            const float p1 = exp2f((s1[r] - mnew) * 1.44269504f);
            float rs = p0 + p1;
#pragma unroll
            for (int off = 1; off < 16; off <<= 1)
                rs += __shfl_xor(rs, off, 32);
            l_r[r] = l_r[r] * alpha[r] + rs;
            m_r[r] = mnew;
            s0[r] = p0; s1[r] = p1;
        }

        // C-layout P -> LDS -> A-layout fragment (16x32 bf16)
#pragma unroll
        for (int r = 0; r < 8; ++r) {
            P[(r + 8 * half) * 32 + mn]      = (__bf16)s0[r];
            P[(r + 8 * half) * 32 + 16 + mn] = (__bf16)s1[r];
        }
        asm volatile("s_wait_dscnt 0" ::: "memory");
        const v16bf pa = frag_bf16(P + mn * 32, 0, half);

        // rescale accumulators, then O += P @ V
#pragma unroll
        for (int t = 0; t < 4; ++t)
#pragma unroll
            for (int r = 0; r < 8; ++r) O[t][r] *= alpha[r];

#pragma unroll
        for (int t = 0; t < 4; ++t) {
            const __bf16* vrow = VldsT + (t * 16 + mn) * 32;
            O[t] = wmma_bf16(pa, frag_bf16(vrow, 0, half), O[t]);
        }
    }

    float inv[8];
#pragma unroll
    for (int r = 0; r < 8; ++r) inv[r] = 1.0f / l_r[r];
#pragma unroll
    for (int r = 0; r < 8; ++r) {
        const int srow = qbase + wave * 16 + r + 8 * half;
        __bf16* cp = CTX + ((size_t)srow * BATCH + b) * EMB + h * HD;
        cp[0 * 16 + mn] = (__bf16)(O[0][r] * inv[r]);
        cp[1 * 16 + mn] = (__bf16)(O[1][r] * inv[r]);
        cp[2 * 16 + mn] = (__bf16)(O[2][r] * inv[r]);
        cp[3 * 16 + mn] = (__bf16)(O[3][r] * inv[r]);
    }
}

// ---- kernel 3: output projection -------------------------------------------
__global__ __launch_bounds__(256) void out_proj_kernel(
    const unsigned short* __restrict__ CTX_, const float* __restrict__ W,
    const float* __restrict__ bias, float* __restrict__ out)
{
    const __bf16* CTX = reinterpret_cast<const __bf16*>(CTX_);

    const int lane = threadIdx.x & 31;
    const int wave = threadIdx.x >> 5;
    const int half = lane >> 4;
    const int mn   = lane & 15;
    const int m0 = blockIdx.x * 64  + (wave & 1) * 32;
    const int n0 = blockIdx.y * 128 + (wave >> 1) * 32;

    const __bf16* arow0 = CTX + (size_t)(m0 + mn) * EMB;
    const __bf16* arow1 = CTX + (size_t)(m0 + 16 + mn) * EMB;
    const float*  brow0 = W   + (size_t)(n0 + mn) * EMB;
    const float*  brow1 = W   + (size_t)(n0 + 16 + mn) * EMB;

    v8f acc00 = {}, acc01 = {}, acc10 = {}, acc11 = {};
    for (int k0 = 0; k0 < EMB; k0 += 32) {
        v16bf a0 = frag_bf16(arow0, k0, half);
        v16bf a1 = frag_bf16(arow1, k0, half);
        v16bf b0 = frag_f32(brow0, k0, half);
        v16bf b1 = frag_f32(brow1, k0, half);
        acc00 = wmma_bf16(a0, b0, acc00);
        acc01 = wmma_bf16(a0, b1, acc01);
        acc10 = wmma_bf16(a1, b0, acc10);
        acc11 = wmma_bf16(a1, b1, acc11);
    }
    const float bv0 = bias[n0 + mn];
    const float bv1 = bias[n0 + 16 + mn];
#pragma unroll
    for (int r = 0; r < 8; ++r) {
        out[(size_t)(m0 + r + 8 * half) * EMB      + n0 + mn]      = acc00[r] + bv0;
        out[(size_t)(m0 + r + 8 * half) * EMB      + n0 + 16 + mn] = acc01[r] + bv1;
        out[(size_t)(m0 + 16 + r + 8 * half) * EMB + n0 + mn]      = acc10[r] + bv0;
        out[(size_t)(m0 + 16 + r + 8 * half) * EMB + n0 + 16 + mn] = acc11[r] + bv1;
    }
}

// ---- launcher ---------------------------------------------------------------
extern "C" void kernel_launch(void* const* d_in, const int* in_sizes, int n_in,
                              void* d_out, int out_size, void* d_ws, size_t ws_size,
                              hipStream_t stream) {
    const float* query = (const float*)d_in[0];
    const float* w_in  = (const float*)d_in[1];
    const float* b_in  = (const float*)d_in[2];
    const float* w_out = (const float*)d_in[3];
    const float* b_out = (const float*)d_in[4];

    unsigned short* Qw  = (unsigned short*)d_ws;       // bf16, SBE elems each
    unsigned short* Kw  = Qw + (size_t)SBE;
    unsigned short* Vw  = Kw + (size_t)SBE;
    unsigned short* CTX = Vw + (size_t)SBE;

    dim3 gA(S_LEN * BATCH / 64, F3 / 128);             // 128 x 24
    qkv_proj_kernel<<<gA, 256, 0, stream>>>(query, w_in, b_in, Qw, Kw, Vw);

    dim3 gB(S_LEN / 64, BATCH * HEADS);                // 32 x 64
    attn_kernel<<<gB, 128, 0, stream>>>(Qw, Kw, Vw, CTX);

    dim3 gC(S_LEN * BATCH / 64, EMB / 128);            // 128 x 8
    out_proj_kernel<<<gC, 256, 0, stream>>>(CTX, w_out, b_out, (float*)d_out);
}
